// SelfAttentionWithUniformRope_54528904790747
// MI455X (gfx1250) — matrix operand order repaired and checked by reading, not compiled, not run on hardware
//
#include <hip/hip_runtime.h>
#include <hip/hip_fp16.h>

#define BATCH 16
#define SEQ   1025
#define SEQP  1040          // padded to multiple of 16
#define EMB   384
#define NH    6
#define HD    64
#define NF    32
#define KTILES 12           // 384/32 k-steps
#define QTILES 65           // ceil(1025/16)
#define KBLOCKS 33          // ceil(1025/32)

typedef __attribute__((ext_vector_type(16))) _Float16 v16h;
typedef __attribute__((ext_vector_type(8)))  float    v8f;

static __device__ __forceinline__ v8f wmma_f16(v16h a, v16h b, v8f c) {
  // D = A(16x32 f16) * B(32x16 f16) + C(16x16 f32)
  return __builtin_amdgcn_wmma_f32_16x16x32_f16(false, a, false, b, (short)0, c, false, false);
}

// A-fragment swizzle: element (m, k) of a 16x32 f16 tile -> (lane, slot)
// slot e in v16h; lane = m + 16*g with g=k>>4, e=k&15
static __device__ __forceinline__ int a_lane(int m, int kk) { return m + 16 * ((kk >> 4) & 1); }
static __device__ __forceinline__ int a_slot(int kk)        { return kk & 15; }

// ---------------- workspace layout (bytes, all 256B aligned) ----------------
static constexpr size_t SZ_TRIG  = (size_t)SEQ * NH * NF * 4;          // 787200
static constexpr size_t SZ_XS    = (size_t)BATCH * SEQ * EMB * 2;      // 12595200 (A-swizzled)
static constexpr size_t SZ_W     = (size_t)EMB * EMB * 2;              // 294912
static constexpr size_t SZ_QS    = (size_t)BATCH * NH * QTILES * 2 * 32 * 16 * 2; // 12779520
static constexpr size_t SZ_KVH   = (size_t)BATCH * NH * SEQP * HD * 2; // 12779520
static constexpr size_t OFF_COS  = 0;
static constexpr size_t OFF_SIN  = OFF_COS + SZ_TRIG;
static constexpr size_t OFF_XS   = OFF_SIN + SZ_TRIG;
static constexpr size_t OFF_WT   = OFF_XS + SZ_XS;      // Wq,Wk,Wv,Wo f16 transposed [n][k]
static constexpr size_t OFF_QS   = OFF_WT + 4 * SZ_W;   // A-swizzled roped Q
static constexpr size_t OFF_KH   = OFF_QS + SZ_QS;      // [bh][sp][d]
static constexpr size_t OFF_VT   = OFF_KH + SZ_KVH;     // [bh][d][sp]
static constexpr size_t OFF_QRAW = OFF_VT + SZ_KVH;
static constexpr size_t OFF_KRAW = OFF_QRAW + SZ_XS;
static constexpr size_t OFF_OS   = OFF_QRAW;            // alias: Qraw dead once RoPE done

// ---------------- 1. RoPE trig tables (matches jax reference) --------------
__global__ void trig_kernel(float* __restrict__ cosT, float* __restrict__ sinT) {
  int idx = blockIdx.x * blockDim.x + threadIdx.x;
  if (idx >= SEQ * NH * NF) return;
  int f = idx % NF;
  int h = (idx / NF) % NH;
  int s = idx / (NF * NH);
  float g = 2.f;
  #pragma unroll
  for (int i = 0; i < 10; ++i) g = cbrtf(1.f + g);   // phi(2)
  float a1 = 1.f / g, a2 = a1 * a1;
  float irow = (float)(h * NF + f + 1);
  float z1 = fmodf(irow * a1, 1.f);
  float z2 = fmodf(irow * a2, 1.f);
  float d1 = erfinvf(2.f * z1 - 1.f);
  float d2 = erfinvf(2.f * z2 - 1.f);
  float inv = rsqrtf(d1 * d1 + d2 * d2);
  float omega = 0.1f * powf(10000.f, (float)f / (float)(NF - 1));
  float fx = d1 * inv * omega;
  float fy = d2 * inv * omega;
  float cx = 0.f, cy = 0.f;
  if (s > 0) {
    int p = s - 1;
    cx = (float)(p & 31) / 31.f * 2.f - 1.f;
    cy = (float)(p >> 5) / 31.f * 2.f - 1.f;
  }
  float th = fx * cx + fy * cy;
  cosT[idx] = cosf(th);
  sinT[idx] = sinf(th);
}

// ---------------- 2a. x: f32 -> f16, A-fragment swizzled -------------------
// xs layout: [mtile(1025)][ktile(12)][lane(32)][slot(16)]
__global__ void cvt_x_kernel(const float* __restrict__ src, _Float16* __restrict__ xs) {
  int i = blockIdx.x * blockDim.x + threadIdx.x;
  if (i >= BATCH * SEQ * EMB) return;
  int row = i / EMB, k = i % EMB;
  int mt = row >> 4, m = row & 15;
  int kt = k >> 5, kk = k & 31;
  size_t dst = (((size_t)mt * KTILES + kt) * 32 + a_lane(m, kk)) * 16 + a_slot(kk);
  xs[dst] = (_Float16)src[i];
}

// ---------------- 2b. W: f32 [k][n] -> f16 transposed [n][k] ---------------
__global__ void cvt_wt_kernel(const float* __restrict__ src, _Float16* __restrict__ wt) {
  int i = blockIdx.x * blockDim.x + threadIdx.x;
  if (i >= EMB * EMB) return;
  int k = i / EMB, n = i % EMB;
  wt[(size_t)n * EMB + k] = (_Float16)src[i];
}

// ---------------- 3. fused QKV GEMM (one wave per 16x16 tile) --------------
__global__ void __launch_bounds__(128) gemm_qkv_kernel(
    const _Float16* __restrict__ xs,
    const _Float16* __restrict__ Wqt, const _Float16* __restrict__ Wkt,
    const _Float16* __restrict__ Wvt,
    _Float16* __restrict__ Qraw, _Float16* __restrict__ Kraw,
    _Float16* __restrict__ Vt) {
  const int lane = threadIdx.x & 31;
  const int wave = (blockIdx.x * blockDim.x + threadIdx.x) >> 5;
  const int NT = EMB / 16;               // 24
  const int MT = (BATCH * SEQ) / 16;     // 1025
  int which = wave / (MT * NT);
  int rem   = wave % (MT * NT);
  int mt = rem / NT, nt = rem % NT;
  const _Float16* Wt = (which == 0) ? Wqt : (which == 1) ? Wkt : Wvt;
  int m0 = mt * 16, n0 = nt * 16;
  int g = lane >> 4, nc = lane & 15;
  const _Float16* arow = xs + ((size_t)mt * KTILES) * 32 * 16 + lane * 16;
  const _Float16* brow = Wt + (size_t)(n0 + nc) * EMB + 16 * g;
  v8f acc = {};
  for (int ks = 0; ks < KTILES; ++ks) {
    v16h a = *(const v16h*)(arow + (size_t)ks * 512);
    v16h b = *(const v16h*)(brow + ks * 32);
    acc = wmma_f16(a, b, acc);
  }
  #pragma unroll
  for (int rr = 0; rr < 8; ++rr) {
    int row = m0 + rr + 8 * g;
    int col = n0 + nc;
    if (which < 2) {
      _Float16* dst = (which == 0) ? Qraw : Kraw;
      dst[(size_t)row * EMB + col] = (_Float16)acc[rr];
    } else {
      int b = row / SEQ, s = row % SEQ;
      int h = col / HD,  d = col % HD;
      Vt[(((size_t)(b * NH + h)) * HD + d) * SEQP + s] = (_Float16)acc[rr];
    }
  }
}

// ---------------- 4. zero padded key-columns of Vt -------------------------
__global__ void zero_vt_kernel(_Float16* Vt) {
  int i = blockIdx.x * blockDim.x + threadIdx.x;
  const int n = BATCH * NH * HD * (SEQP - SEQ);   // 92160
  if (i >= n) return;
  int s   = i % (SEQP - SEQ);
  int bhd = i / (SEQP - SEQ);
  Vt[(size_t)bhd * SEQP + SEQ + s] = (_Float16)0.f;
}

// ---------------- 5. RoPE + repack; Q pre-scaled by 1/8, A-swizzled --------
__global__ void rope_kernel(const _Float16* __restrict__ Qraw,
                            const _Float16* __restrict__ Kraw,
                            const float* __restrict__ cosT,
                            const float* __restrict__ sinT,
                            _Float16* __restrict__ Qs, _Float16* __restrict__ Kh) {
  int idx = blockIdx.x * blockDim.x + threadIdx.x;
  if (idx >= BATCH * SEQ * NH * NF) return;
  int f  = idx % NF;
  int h  = (idx / NF) % NH;
  int bs = idx / (NF * NH);
  int b = bs / SEQ, s = bs % SEQ;
  int bh = b * NH + h;
  float c  = cosT[(s * NH + h) * NF + f];
  float sn = sinT[(s * NH + h) * NF + f];
  size_t src = (size_t)bs * EMB + h * HD + f;
  int qt = s >> 4, m = s & 15;
  // Q -> A-swizzled [bh][qt][fr][lane][slot]
  {
    float x1 = (float)Qraw[src], y1 = (float)Qraw[src + NF];
    float q0 = 0.125f * (x1 * c - y1 * sn);       // d = f      (fr = f>>5 = 0 when f<32)
    float q1 = 0.125f * (x1 * sn + y1 * c);       // d = f + 32
    int d0 = f, d1 = f + NF;
    size_t base = (((size_t)bh * QTILES + qt) * 2) * 512;
    size_t i0 = base + (size_t)(d0 >> 5) * 512 + a_lane(m, d0 & 31) * 16 + a_slot(d0 & 31);
    size_t i1 = base + (size_t)(d1 >> 5) * 512 + a_lane(m, d1 & 31) * 16 + a_slot(d1 & 31);
    Qs[i0] = (_Float16)q0;
    Qs[i1] = (_Float16)q1;
  }
  // K -> [bh][sp][d]
  {
    float x1 = (float)Kraw[src], y1 = (float)Kraw[src + NF];
    size_t dst = ((size_t)bh * SEQP + s) * HD + f;
    Kh[dst]      = (_Float16)(x1 * c - y1 * sn);
    Kh[dst + NF] = (_Float16)(x1 * sn + y1 * c);
  }
}

// ---------------- 6. streaming-softmax attention (1 wave / 16 q-rows) ------
__global__ void __launch_bounds__(128) attn_kernel(
    const _Float16* __restrict__ Qs, const _Float16* __restrict__ Kh,
    const _Float16* __restrict__ Vt, _Float16* __restrict__ Os) {
  __shared__ __align__(32) _Float16 ldsP[4][16 * 32];   // P-tile transpose bounce per wave
  const int lane = threadIdx.x & 31;
  const int wib  = threadIdx.x >> 5;
  const int wave = (blockIdx.x * blockDim.x + threadIdx.x) >> 5;
  int qt = wave % QTILES;
  int bh = wave / QTILES;
  int h = bh % NH, b = bh / NH;
  const _Float16* Kb = Kh + (size_t)bh * SEQP * HD;
  const _Float16* Vb = Vt + (size_t)bh * HD * SEQP;
  int m0 = qt * 16;
  int mrow = lane & 15, g = lane >> 4, nc = lane & 15;

  // Q tile: two contiguous A-fragments
  v16h aq[2];
  {
    const _Float16* p = Qs + (((size_t)bh * QTILES + qt) * 2) * 512 + lane * 16;
    aq[0] = *(const v16h*)(p);
    aq[1] = *(const v16h*)(p + 512);
  }

  v8f oacc[4] = {v8f{}, v8f{}, v8f{}, v8f{}};
  float mrun[8], lrun[8];
  #pragma unroll
  for (int rr = 0; rr < 8; ++rr) { mrun[rr] = -1e30f; lrun[rr] = 0.f; }

  for (int kb = 0; kb < KBLOCKS; ++kb) {
    int key0 = kb * 32;
    // S = Q * K^T  (K rows are exactly B lane-stripes)
    v8f s0 = {}, s1 = {};
    #pragma unroll
    for (int fr = 0; fr < 2; ++fr) {
      v16h bk = *(const v16h*)(Kb + (size_t)(key0 + nc) * HD + fr * 32 + 16 * g);
      s0 = wmma_f16(aq[fr], bk, s0);
    }
    #pragma unroll
    for (int fr = 0; fr < 2; ++fr) {
      v16h bk = *(const v16h*)(Kb + (size_t)(key0 + 16 + nc) * HD + fr * 32 + 16 * g);
      s1 = wmma_f16(aq[fr], bk, s1);
    }
    bool v0 = (key0 + nc) < SEQ;
    bool v1 = (key0 + 16 + nc) < SEQ;
    // online softmax per output row (rows live across 16-lane half-waves)
    #pragma unroll
    for (int rr = 0; rr < 8; ++rr) {
      float e0 = v0 ? s0[rr] : -1e30f;
      float e1 = v1 ? s1[rr] : -1e30f;
      float bm = fmaxf(e0, e1);
      #pragma unroll
      for (int off = 1; off < 16; off <<= 1) bm = fmaxf(bm, __shfl_xor(bm, off, 32));
      float mnew  = fmaxf(mrun[rr], bm);
      float scale = __expf(mrun[rr] - mnew);
      float p0 = __expf(e0 - mnew);
      float p1 = __expf(e1 - mnew);
      float rs = p0 + p1;
      #pragma unroll
      for (int off = 1; off < 16; off <<= 1) rs += __shfl_xor(rs, off, 32);
      lrun[rr] = lrun[rr] * scale + rs;
      mrun[rr] = mnew;
      oacc[0][rr] *= scale; oacc[1][rr] *= scale;
      oacc[2][rr] *= scale; oacc[3][rr] *= scale;
      int row = rr + 8 * g;
      ldsP[wib][row * 32 + nc]      = (_Float16)p0;
      ldsP[wib][row * 32 + 16 + nc] = (_Float16)p1;
    }
    asm volatile("s_wait_dscnt 0x0" ::: "memory");
    // reload P (C-layout -> A-layout) via LDS
    v16h aP;
    {
      const _Float16* lp = &ldsP[wib][mrow * 32];
      #pragma unroll
      for (int e = 0; e < 8; ++e) {
        aP[e]     = lp[16 * g + e];
        aP[8 + e] = lp[16 * g + 8 + e];
      }
    }
    // O += P * V   (V transposed: contiguous B lane-stripes)
    #pragma unroll
    for (int t = 0; t < 4; ++t) {
      v16h bv = *(const v16h*)(Vb + (size_t)(t * 16 + nc) * SEQP + key0 + 16 * g);
      oacc[t] = wmma_f16(aP, bv, oacc[t]);
    }
  }
  // normalize + store into A-swizzled layout for the output GEMM
  #pragma unroll
  for (int rr = 0; rr < 8; ++rr) {
    float inv = 1.f / lrun[rr];
    int s = m0 + rr + 8 * g;
    if (s < SEQ) {
      int row = b * SEQ + s;
      int mt = row >> 4, m = row & 15;
      #pragma unroll
      for (int t = 0; t < 4; ++t) {
        int col = h * HD + t * 16 + nc;
        int kt = col >> 5, kk = col & 31;
        size_t dst = (((size_t)mt * KTILES + kt) * 32 + a_lane(m, kk)) * 16 + a_slot(kk);
        Os[dst] = (_Float16)(oacc[t][rr] * inv);
      }
    }
  }
}

// ---------------- 7. output projection + bias -> f32 -----------------------
__global__ void __launch_bounds__(128) gemm_out_kernel(
    const _Float16* __restrict__ Os, const _Float16* __restrict__ Wot,
    const float* __restrict__ bo, float* __restrict__ out) {
  const int lane = threadIdx.x & 31;
  const int wave = (blockIdx.x * blockDim.x + threadIdx.x) >> 5;
  const int NT = EMB / 16;
  int mt = wave / NT, nt = wave % NT;
  int m0 = mt * 16, n0 = nt * 16;
  int g = lane >> 4, nc = lane & 15;
  const _Float16* arow = Os + ((size_t)mt * KTILES) * 32 * 16 + lane * 16;
  const _Float16* brow = Wot + (size_t)(n0 + nc) * EMB + 16 * g;
  v8f acc = {};
  for (int ks = 0; ks < KTILES; ++ks) {
    v16h a = *(const v16h*)(arow + (size_t)ks * 512);
    v16h b = *(const v16h*)(brow + ks * 32);
    acc = wmma_f16(a, b, acc);
  }
  float bias = bo[n0 + nc];
  #pragma unroll
  for (int rr = 0; rr < 8; ++rr) {
    int row = m0 + rr + 8 * g;
    out[(size_t)row * EMB + n0 + nc] = acc[rr] + bias;
  }
}

// ---------------------------------------------------------------------------
extern "C" void kernel_launch(void* const* d_in, const int* in_sizes, int n_in,
                              void* d_out, int out_size, void* d_ws, size_t ws_size,
                              hipStream_t stream) {
  const float* x  = (const float*)d_in[0];
  const float* Wq = (const float*)d_in[1];
  const float* Wk = (const float*)d_in[2];
  const float* Wv = (const float*)d_in[3];
  const float* Wo = (const float*)d_in[4];
  const float* bo = (const float*)d_in[5];

  char* ws = (char*)d_ws;
  float*     cosT = (float*)(ws + OFF_COS);
  float*     sinT = (float*)(ws + OFF_SIN);
  _Float16*  xs   = (_Float16*)(ws + OFF_XS);
  _Float16*  Wqt  = (_Float16*)(ws + OFF_WT + 0 * SZ_W);
  _Float16*  Wkt  = (_Float16*)(ws + OFF_WT + 1 * SZ_W);
  _Float16*  Wvt  = (_Float16*)(ws + OFF_WT + 2 * SZ_W);
  _Float16*  Wot  = (_Float16*)(ws + OFF_WT + 3 * SZ_W);
  _Float16*  Qs   = (_Float16*)(ws + OFF_QS);
  _Float16*  Kh   = (_Float16*)(ws + OFF_KH);
  _Float16*  Vt   = (_Float16*)(ws + OFF_VT);
  _Float16*  Qraw = (_Float16*)(ws + OFF_QRAW);
  _Float16*  Kraw = (_Float16*)(ws + OFF_KRAW);
  _Float16*  Os   = (_Float16*)(ws + OFF_OS);

  // 1. trig tables
  {
    int n = SEQ * NH * NF;
    trig_kernel<<<(n + 255) / 256, 256, 0, stream>>>(cosT, sinT);
  }
  // 2. converts (x swizzled, weights transposed)
  {
    int n = BATCH * SEQ * EMB;
    cvt_x_kernel<<<(n + 255) / 256, 256, 0, stream>>>(x, xs);
    int m = EMB * EMB;
    cvt_wt_kernel<<<(m + 255) / 256, 256, 0, stream>>>(Wq, Wqt);
    cvt_wt_kernel<<<(m + 255) / 256, 256, 0, stream>>>(Wk, Wkt);
    cvt_wt_kernel<<<(m + 255) / 256, 256, 0, stream>>>(Wv, Wvt);
    cvt_wt_kernel<<<(m + 255) / 256, 256, 0, stream>>>(Wo, Wot);
  }
  // 3. QKV GEMM: 3 * 1025 * 24 tiles, 4 waves/block
  {
    int waves = 3 * ((BATCH * SEQ) / 16) * (EMB / 16);   // 73800
    gemm_qkv_kernel<<<waves / 4, 128, 0, stream>>>(xs, Wqt, Wkt, Wvt, Qraw, Kraw, Vt);
  }
  // 4. zero padded key columns of Vt (0 * NaN protection)
  {
    int n = BATCH * NH * HD * (SEQP - SEQ);
    zero_vt_kernel<<<(n + 255) / 256, 256, 0, stream>>>(Vt);
  }
  // 5. RoPE + repack
  {
    int n = BATCH * SEQ * NH * NF;
    rope_kernel<<<(n + 255) / 256, 256, 0, stream>>>(Qraw, Kraw, cosT, sinT, Qs, Kh);
  }
  // 6. attention: 16*6*65 = 6240 waves, 4 waves/block
  {
    int waves = BATCH * NH * QTILES;
    attn_kernel<<<waves / 4, 128, 0, stream>>>(Qs, Kh, Vt, Os);
  }
  // 7. output projection: 1025 * 24 tiles
  {
    int waves = ((BATCH * SEQ) / 16) * (EMB / 16);       // 24600
    gemm_out_kernel<<<waves / 4, 128, 0, stream>>>(Os, Wot, bo, (float*)d_out);
  }
}
